// SparseWeightEnsemblingMoE_10222022164653
// MI455X (gfx1250) — matrix operand order repaired and checked by reading, compiled, not run on hardware
//
#include <hip/hip_runtime.h>
#include <hip/hip_bf16.h>

// ---------------------------------------------------------------------------
// Problem sizes (fixed by the reference)
// ---------------------------------------------------------------------------
#define SEQ   2048
#define BATCH 4
#define MDIM  (SEQ * BATCH)   // 8192 rows
#define HDIM  1024
#define FDIM  4096
#define EDIM  8

typedef __bf16 bf16;
typedef __attribute__((ext_vector_type(16))) __bf16 v16bf;
typedef __attribute__((ext_vector_type(8)))  float  v8f;
typedef __attribute__((ext_vector_type(4)))  __bf16 v4bf;
typedef __attribute__((ext_vector_type(4)))  int    v4i;

union Frag {
    v16bf  v;
    float4 f[2];
};

// ---------------------------------------------------------------------------
// CDNA5 async global->LDS copy (ASYNCcnt path), with sync fallback
// ---------------------------------------------------------------------------
#if __has_builtin(__builtin_amdgcn_global_load_async_to_lds_b128)
#define HAVE_ASYNC_LDS 1
#else
#define HAVE_ASYNC_LDS 0
#endif

typedef __attribute__((address_space(1))) v4i* gas_v4i_ptr;
typedef __attribute__((address_space(3))) v4i* las_v4i_ptr;

__device__ __forceinline__ void copy16_to_lds(void* lds, const void* g) {
#if HAVE_ASYNC_LDS
    __builtin_amdgcn_global_load_async_to_lds_b128(
        (gas_v4i_ptr)(uintptr_t)g,
        (las_v4i_ptr)(uintptr_t)lds,
        0, 0);
#else
    *reinterpret_cast<float4*>(lds) = *reinterpret_cast<const float4*>(g);
#endif
}

template <int N>
__device__ __forceinline__ void wait_async() {
#if HAVE_ASYNC_LDS
#if __has_builtin(__builtin_amdgcn_s_wait_asynccnt)
    __builtin_amdgcn_s_wait_asynccnt(N);
#else
    asm volatile("s_wait_asynccnt %0" :: "i"(N) : "memory");
#endif
#endif
}

// ---------------------------------------------------------------------------
// Elementwise helpers
// ---------------------------------------------------------------------------
__global__ void f32_to_bf16_kernel(const float* __restrict__ in,
                                   bf16* __restrict__ out, int n4) {
    int i = blockIdx.x * blockDim.x + threadIdx.x;
    if (i < n4) {
        float4 v = reinterpret_cast<const float4*>(in)[i];
        v4bf o;
        o[0] = (bf16)v.x; o[1] = (bf16)v.y; o[2] = (bf16)v.z; o[3] = (bf16)v.w;
        reinterpret_cast<v4bf*>(out)[i] = o;
    }
}

// merged weight = bf16(base + sum_e lam[e] * tv[e]); float4 streaming
__global__ void merge_w_kernel(const float* __restrict__ base,
                               const float* __restrict__ tv,
                               const float* __restrict__ lam,
                               bf16* __restrict__ out,
                               int n4, int estride4) {
    int i = blockIdx.x * blockDim.x + threadIdx.x;
    if (i >= n4) return;
    float l[EDIM];
#pragma unroll
    for (int e = 0; e < EDIM; ++e) l[e] = lam[e];
    float4 acc = reinterpret_cast<const float4*>(base)[i];
#pragma unroll
    for (int e = 0; e < EDIM; ++e) {
        float4 t = reinterpret_cast<const float4*>(tv)[(size_t)e * estride4 + i];
        acc.x += l[e] * t.x; acc.y += l[e] * t.y;
        acc.z += l[e] * t.z; acc.w += l[e] * t.w;
    }
    v4bf o;
    o[0] = (bf16)acc.x; o[1] = (bf16)acc.y; o[2] = (bf16)acc.z; o[3] = (bf16)acc.w;
    reinterpret_cast<v4bf*>(out)[i] = o;
}

__global__ void merge_b_kernel(const float* __restrict__ base,
                               const float* __restrict__ tv,
                               const float* __restrict__ lam,
                               float* __restrict__ out,
                               int n, int estride) {
    int i = blockIdx.x * blockDim.x + threadIdx.x;
    if (i >= n) return;
    float acc = base[i];
#pragma unroll
    for (int e = 0; e < EDIM; ++e) acc += lam[e] * tv[(size_t)e * estride + i];
    out[i] = acc;
}

// deterministic fold of per-Mtile gate partials -> gmean[HDIM]
__global__ void reduce_partials_kernel(const float* __restrict__ partial,
                                       float* __restrict__ gmean,
                                       int mtiles, int n) {
    int col = blockIdx.x * blockDim.x + threadIdx.x;
    if (col >= n) return;
    float s = 0.f;
    for (int mt = 0; mt < mtiles; ++mt) s += partial[(size_t)mt * n + col];
    gmean[col] = s;
}

// lam[e] = gate_b2[e] + (1/M) * dot(gmean, gate_w2[e,:])  (8 waves, one e each)
__global__ void gate_lambda_kernel(const float* __restrict__ gmean,
                                   const float* __restrict__ gw2,
                                   const float* __restrict__ gb2,
                                   float* __restrict__ lam) {
    int w    = threadIdx.x >> 5;   // wave id == expert id
    int lane = threadIdx.x & 31;
    float s = 0.f;
    for (int h = lane; h < HDIM; h += 32) s += gmean[h] * gw2[w * HDIM + h];
#pragma unroll
    for (int off = 16; off > 0; off >>= 1) s += __shfl_xor(s, off, 32);
    if (lane == 0 && w < EDIM) lam[w] = gb2[w] + s * (1.0f / (float)MDIM);
}

// ---------------------------------------------------------------------------
// Tiled bf16 WMMA GEMM:  D[m,n] = sum_k A[m,k] * B[n,k]  (+ bias[n])
//   block tile: 128 (M) x 128 (N), K-step 32, 256 threads = 8 waves
//   wave (wm 0..1, wn 0..3) owns a 64x32 macro tile = 4x2 WMMA fragments
//   double-buffered LDS staged with GLOBAL_LOAD_ASYNC_TO_LDS_B128 (ASYNCcnt)
// MODE 0: relu(.+bias) -> bf16 store       (FFN fc1)
// MODE 1: (.+bias)     -> f32 store        (FFN fc2)
// MODE 2: relu(.+bias) -> per-block column sums -> partial[Mtile][Ntot] (gate)
// ---------------------------------------------------------------------------
#define LDST 40   // padded LDS row stride (bf16 elems): 80B rows, 16B aligned

template <int MODE>
__global__ __launch_bounds__(256)
void wmma_gemm_kernel(const bf16* __restrict__ A, const bf16* __restrict__ B,
                      const float* __restrict__ bias, int K, int Ntot,
                      bf16* __restrict__ outbf, float* __restrict__ outf) {
    __shared__ alignas(16) bf16 As[2][128 * LDST];   // 2 x 10 KB
    __shared__ alignas(16) bf16 Bs[2][128 * LDST];   // 2 x 10 KB
    __shared__ float red[2][128];                    // gate-mode reduction

    const int tid  = threadIdx.x;
    const int lane = tid & 31;
    const int half = lane >> 4;    // 0: lanes 0-15, 1: lanes 16-31
    const int lm   = lane & 15;
    const int wave = tid >> 5;
    const int wm   = wave & 1;     // 64-row slice of the 128-row block tile
    const int wn   = wave >> 1;    // 32-col slice of the 128-col block tile

    const int m0 = blockIdx.y * 128;
    const int n0 = blockIdx.x * 128;

    v8f acc[4][2] = {};

    // stage one 128x32 A tile + 128x32 B tile (16 KB): 1024 x 16B chunks,
    // 4 per thread -> 4 async-load instructions per wave per tile
    auto stage = [&](int buf, int k0) {
#pragma unroll
        for (int c = 0; c < 2; ++c) {
            int chunk = tid + c * 256;
            int row = chunk >> 2, seg = chunk & 3;
            copy16_to_lds(&As[buf][row * LDST + seg * 8],
                          A + (size_t)(m0 + row) * K + k0 + seg * 8);
        }
#pragma unroll
        for (int c = 0; c < 2; ++c) {
            int chunk = tid + c * 256;
            int row = chunk >> 2, seg = chunk & 3;
            copy16_to_lds(&Bs[buf][row * LDST + seg * 8],
                          B + (size_t)(n0 + row) * K + k0 + seg * 8);
        }
    };

    stage(0, 0);
    const int nsteps = K >> 5;
    for (int it = 0; it < nsteps; ++it) {
        const int cur = it & 1;
        if (it + 1 < nsteps) {
            stage(cur ^ 1, (it + 1) << 5);   // prefetch next tile
            wait_async<4>();                 // drain current tile's 4 loads
        } else {
            wait_async<0>();
        }
        __syncthreads();   // all waves' current-tile data visible in LDS

        // A fragment (16x32): elem e<8 -> K=8*half+e ; e>=8 -> K=16+8*half+(e-8)
        Frag a[4], b[2];
#pragma unroll
        for (int t = 0; t < 4; ++t) {
            const bf16* p = &As[cur][(wm * 64 + t * 16 + lm) * LDST];
            a[t].f[0] = *reinterpret_cast<const float4*>(p + 8 * half);
            a[t].f[1] = *reinterpret_cast<const float4*>(p + 16 + 8 * half);
        }
        // B fragment (32x16): elem e<8 -> K=8*half+e ; e>=8 -> K=16+8*half+(e-8), col = lm
#pragma unroll
        for (int u = 0; u < 2; ++u) {
            const bf16* p = &Bs[cur][(wn * 32 + u * 16 + lm) * LDST];
            b[u].f[0] = *reinterpret_cast<const float4*>(p + 8 * half);
            b[u].f[1] = *reinterpret_cast<const float4*>(p + 16 + 8 * half);
        }
#pragma unroll
        for (int t = 0; t < 4; ++t)
#pragma unroll
            for (int u = 0; u < 2; ++u)
                acc[t][u] = __builtin_amdgcn_wmma_f32_16x16x32_bf16(
                    false, a[t].v, false, b[u].v, (short)0, acc[t][u], false, false);

        __syncthreads();   // everyone done reading 'cur' before it is re-staged
    }

    // epilogue: D layout is M = r + 8*half, N = lm  (per 16x16 tile)
#pragma unroll
    for (int u = 0; u < 2; ++u) {
        const int col = n0 + wn * 32 + u * 16 + lm;
        const float bn = bias[col];
        if (MODE == 2) {
            float s = 0.f;
#pragma unroll
            for (int t = 0; t < 4; ++t)
#pragma unroll
                for (int r = 0; r < 8; ++r)
                    s += fmaxf(acc[t][u][r] + bn, 0.f);
            s += __shfl_xor(s, 16, 32);                  // fold lane halves
            if (half == 0) red[wm][wn * 32 + u * 16 + lm] = s;
        } else {
#pragma unroll
            for (int t = 0; t < 4; ++t)
#pragma unroll
                for (int r = 0; r < 8; ++r) {
                    int row = m0 + wm * 64 + t * 16 + half * 8 + r;
                    float v = acc[t][u][r] + bn;
                    if (MODE == 0) {
                        outbf[(size_t)row * Ntot + col] = (bf16)fmaxf(v, 0.f);
                    } else {
                        outf[(size_t)row * Ntot + col] = v;
                    }
                }
        }
    }
    if (MODE == 2) {
        __syncthreads();
        if (tid < 128) {
            float s = red[0][tid] + red[1][tid];
            outf[(size_t)blockIdx.y * Ntot + n0 + tid] = s;   // partial[Mtile][col]
        }
    }
}

// ---------------------------------------------------------------------------
// Host-side orchestration
// ---------------------------------------------------------------------------
extern "C" void kernel_launch(void* const* d_in, const int* in_sizes, int n_in,
                              void* d_out, int out_size, void* d_ws, size_t ws_size,
                              hipStream_t stream) {
    const float* hs      = (const float*)d_in[0];   // [S,B,H] -> [M,H]
    const float* gate_w1 = (const float*)d_in[1];   // [H,H]
    const float* gate_b1 = (const float*)d_in[2];   // [H]
    const float* gate_w2 = (const float*)d_in[3];   // [E,H]
    const float* gate_b2 = (const float*)d_in[4];   // [E]
    const float* base_w1 = (const float*)d_in[5];   // [F,H]
    const float* base_b1 = (const float*)d_in[6];   // [F]
    const float* base_w2 = (const float*)d_in[7];   // [H,F]
    const float* base_b2 = (const float*)d_in[8];   // [H]
    const float* tv_w1   = (const float*)d_in[9];   // [E,F,H]
    const float* tv_b1   = (const float*)d_in[10];  // [E,F]
    const float* tv_w2   = (const float*)d_in[11];  // [E,H,F]
    const float* tv_b2   = (const float*)d_in[12];  // [E,H]

    // workspace carve-up (256B aligned)
    char* p = (char*)d_ws;
    auto alloc = [&](size_t bytes) {
        char* r = p;
        p += (bytes + 255) & ~(size_t)255;
        return r;
    };
    bf16*  Xbf      = (bf16*) alloc((size_t)MDIM * HDIM * sizeof(bf16));   // 16.8 MB
    bf16*  GW1bf    = (bf16*) alloc((size_t)HDIM * HDIM * sizeof(bf16));   //  2.1 MB
    bf16*  W1bf     = (bf16*) alloc((size_t)FDIM * HDIM * sizeof(bf16));   //  8.4 MB
    bf16*  W2bf     = (bf16*) alloc((size_t)HDIM * FDIM * sizeof(bf16));   //  8.4 MB
    bf16*  Hbf      = (bf16*) alloc((size_t)MDIM * FDIM * sizeof(bf16));   // 67.1 MB
    float* partials = (float*)alloc((size_t)(MDIM / 128) * HDIM * sizeof(float)); // 256 KB
    float* gmean    = (float*)alloc(HDIM * sizeof(float));
    float* lam      = (float*)alloc(EDIM * sizeof(float));
    float* b1m      = (float*)alloc(FDIM * sizeof(float));
    float* b2m      = (float*)alloc(HDIM * sizeof(float));

    // 1) fp32 -> bf16 conversions
    {
        int n4 = (MDIM * HDIM) / 4;
        f32_to_bf16_kernel<<<(n4 + 255) / 256, 256, 0, stream>>>(hs, Xbf, n4);
        int g4 = (HDIM * HDIM) / 4;
        f32_to_bf16_kernel<<<(g4 + 255) / 256, 256, 0, stream>>>(gate_w1, GW1bf, g4);
    }

    // 2) gate GEMM: relu(X @ gate_w1^T + b1), column-summed into partials
    {
        dim3 grid(HDIM / 128, MDIM / 128);   // (8, 64)
        wmma_gemm_kernel<2><<<grid, 256, 0, stream>>>(
            Xbf, GW1bf, gate_b1, HDIM, HDIM, nullptr, partials);
        reduce_partials_kernel<<<(HDIM + 255) / 256, 256, 0, stream>>>(
            partials, gmean, MDIM / 128, HDIM);
        gate_lambda_kernel<<<1, 256, 0, stream>>>(gmean, gate_w2, gate_b2, lam);
    }

    // 3) weight / bias merge (bandwidth-bound; streams the 268 MB of tv data)
    {
        int w4 = (FDIM * HDIM) / 4;
        merge_w_kernel<<<(w4 + 255) / 256, 256, 0, stream>>>(base_w1, tv_w1, lam, W1bf, w4, w4);
        merge_w_kernel<<<(w4 + 255) / 256, 256, 0, stream>>>(base_w2, tv_w2, lam, W2bf, w4, w4);
        merge_b_kernel<<<(FDIM + 255) / 256, 256, 0, stream>>>(base_b1, tv_b1, lam, b1m, FDIM, FDIM);
        merge_b_kernel<<<(HDIM + 255) / 256, 256, 0, stream>>>(base_b2, tv_b2, lam, b2m, HDIM, HDIM);
    }

    // 4) FFN fc1: Hbf = bf16(relu(X @ w1m^T + b1m))   [8192 x 4096]
    {
        dim3 grid(FDIM / 128, MDIM / 128);   // (32, 64)
        wmma_gemm_kernel<0><<<grid, 256, 0, stream>>>(
            Xbf, W1bf, b1m, HDIM, FDIM, Hbf, nullptr);
    }

    // 5) FFN fc2: out = Hbf @ w2m^T + b2m   [8192 x 1024], fp32
    {
        dim3 grid(HDIM / 128, MDIM / 128);   // (8, 64)
        wmma_gemm_kernel<1><<<grid, 256, 0, stream>>>(
            Hbf, W2bf, b2m, FDIM, HDIM, nullptr, (float*)d_out);
    }
}
